// BoxEmb_6871947673995
// MI455X (gfx1250) — matrix operand order repaired and checked by reading, not compiled, not run on hardware
//
#include <hip/hip_runtime.h>
#include <hip/hip_bf16.h>

// ---------------------------------------------------------------------------
// Types / constants
// ---------------------------------------------------------------------------
typedef _Float16 f16;
typedef __attribute__((ext_vector_type(16))) _Float16 v16h;
typedef __attribute__((ext_vector_type(8)))  _Float16 v8h;
typedef __attribute__((ext_vector_type(8)))  float    v8f;

#define N_NODES 2048
#define N_EDGES 32768
#define T_LEN   65
#define LRELU_K 0.01f
#define EPS_K   1e-8f

__constant__ int c_att_in[6]   = {8, 8, 16, 32, 64, 64};
__constant__ int c_att_out[6]  = {8, 16, 32, 64, 64, 128};
__constant__ int c_attg_in[6]  = {2, 8, 16, 32, 64, 64};
__constant__ int c_pool[6]     = {1, 2, 1, 2, 1, 1};
__constant__ int c_prop[7]     = {4, 8, 16, 32, 32, 32, 4};
__constant__ int c_gate[7]     = {1, 8, 16, 32, 32, 32, 1};
// padded K (= round32(Cin*5)) per layer, for the zero-padded weight tensors
__constant__ int c_kp_att[6]   = {64, 64, 96, 160, 320, 320};
__constant__ int c_kp_attg[6]  = {32, 64, 96, 160, 320, 320};
__constant__ int c_kp_prop[6]  = {32, 64, 96, 160, 160, 160};
__constant__ int c_kp_gate[6]  = {32, 64, 96, 160, 160, 160};

__device__ __forceinline__ float actf(float x, int act) {
  if (act == 1) return x > 0.f ? x : LRELU_K * x;              // leaky relu
  if (act == 2) return 1.f / (1.f + __expf(-x));               // sigmoid
  return x;
}

#define WMMA16(a, b, c) \
  __builtin_amdgcn_wmma_f32_16x16x32_f16(false, (a), false, (b), (short)0, (c), false, false)

// ---------------------------------------------------------------------------
// WMMA fragment builders (f16 16x16x32, wave32; A/B share contiguous k order, C/D per CDNA5 ISA 7.12.2)
// A (16x32, MxK): lane m=L%16; half i -> K = kb + i (i=0..15),
//                 kb = k0 + (L>=16 ? 16 : 0)  => one contiguous 16-half run
// B (32x16, KxN): lane n=L%16; half i -> K = k0 + (L>=16?16:0) + i
// C/D (16x16 f32): vgpr v -> M = m0 + v + (L>=16?8:0), N = n0 + L%16
// ---------------------------------------------------------------------------

// A from zero-padded weights [Mp x Kp], unconditional aligned vector loads
__device__ __forceinline__ v16h fragA_pad(const f16* W, int Kp, int m0, int k0, int lane) {
  int m  = m0 + (lane & 15);
  int kb = k0 + ((lane & 16) ? 16 : 0);
  const v8h* p = (const v8h*)(W + (size_t)m * Kp + kb);
  v8h lo = p[0], hi = p[1];
  v16h a;
  #pragma unroll
  for (int i = 0; i < 8; ++i) { a[i] = lo[i]; a[i + 8] = hi[i]; }
  return a;
}

// implicit-GEMM im2col fragment (unguarded; K-pad rows in LDS are zeroed,
// out-of-range columns only feed discarded C columns)
__device__ __forceinline__ v16h fragB_im2col(const f16* buf, int Tin, int k0, int n0,
                                             int lane) {
  v16h b;
  int n  = n0 + (lane & 15);
  int kb = k0 + ((lane & 16) ? 16 : 0);
  #pragma unroll
  for (int i = 0; i < 16; ++i) {
    int kk = kb + i;
    int ci = kk / 5, r = kk - ci * 5;
    b[i] = buf[ci * Tin + n + r];
  }
  return b;
}

// zero padding region [from, to) of an LDS buffer (lane-parallel)
__device__ __forceinline__ void zpad(f16* buf, int from, int to, int lane) {
  for (int i = from + lane; i < to; i += 32) buf[i] = (f16)0.0f;
}

// conv1d VALID, K=5, via WMMA; f16 in (LDS, zero-padded rows), f16 out (LDS)
__device__ __forceinline__ void conv_edge(const f16* W, int Kp, const float* bias,
                                          const f16* in, f16* out, int Cout, int Tin,
                                          int act, int lane) {
  int Tout = Tin - 4;
  for (int m0 = 0; m0 < Cout; m0 += 16) {
    for (int n0 = 0; n0 < Tout; n0 += 16) {
      v8f acc = {};
      for (int k0 = 0; k0 < Kp; k0 += 32) {
        v16h a = fragA_pad(W, Kp, m0, k0, lane);
        v16h b = fragB_im2col(in, Tin, k0, n0, lane);
        acc = WMMA16(a, b, acc);
      }
      int n = n0 + (lane & 15);
      if (n < Tout) {
        #pragma unroll
        for (int v = 0; v < 8; ++v) {
          int m = m0 + v + ((lane & 16) ? 8 : 0);
          int mb = m < Cout ? m : Cout - 1;              // clamped bias read
          out[m * Tout + n] = (f16)actf(acc[v] + bias[mb], act);
        }
      }
    }
  }
}

// conv1d with edge padding already applied (Tin=69 -> Tout=65), f32 out + residual
__device__ __forceinline__ void conv_node(const f16* W, int Kp, const float* bias,
                                          const f16* in, float* out, const float* prev,
                                          int Cout, int act, int lane) {
  const int Tin = 69, Tout = 65;
  for (int m0 = 0; m0 < Cout; m0 += 16) {
    for (int n0 = 0; n0 < Tout; n0 += 16) {
      v8f acc = {};
      for (int k0 = 0; k0 < Kp; k0 += 32) {
        v16h a = fragA_pad(W, Kp, m0, k0, lane);
        v16h b = fragB_im2col(in, Tin, k0, n0, lane);
        acc = WMMA16(a, b, acc);
      }
      int n = n0 + (lane & 15);
      if (n < Tout) {
        #pragma unroll
        for (int v = 0; v < 8; ++v) {
          int m = m0 + v + ((lane & 16) ? 8 : 0);
          int mb = m < Cout ? m : Cout - 1;
          float x = actf(acc[v] + bias[mb], act);
          if (prev) x += prev[m * Tout + n];
          out[m * Tout + n] = x;
        }
      }
    }
  }
}

// ---------------------------------------------------------------------------
// Edge similarity CNN: one wave per edge, all 6 conv layers in LDS via WMMA
// ---------------------------------------------------------------------------
#define ESIM_WAVES 4
#define EBUF 1312   // max C*T intermediate (64*20=1280) + slack

struct EdgeSimW {
  const f16*   aw[6]; const float* ab[6];
  const f16*   gw[6]; const float* gb[6];
  const float* linw;  const float* linb;
};

__global__ void __launch_bounds__(128)
edge_sim_kernel(const float* __restrict__ X, const float* __restrict__ S,
                const int* __restrict__ eidx, float* __restrict__ sim, EdgeSimW w) {
  __shared__ f16 sm[ESIM_WAVES * 4 * EBUF];
  int wid = threadIdx.x >> 5, lane = threadIdx.x & 31;
  f16* xa = sm + (wid * 4 + 0) * EBUF;
  f16* xb = sm + (wid * 4 + 1) * EBUF;
  f16* sa = sm + (wid * 4 + 2) * EBUF;
  f16* sb = sm + (wid * 4 + 3) * EBUF;
  int e  = blockIdx.x * ESIM_WAVES + wid;
  int e0 = eidx[2 * e], e1 = eidx[2 * e + 1];
  if (lane == 0) __builtin_prefetch(w.aw[0], 0, 3);

  // inputs: X_in[0..3]=X[e0]*S[e0], [4..7]=X[e1]*S[e1]; S_in=[S[e0];S[e1]]
  for (int idx = lane; idx < 8 * T_LEN; idx += 32) {
    int ci = idx / T_LEN, t = idx - ci * T_LEN;
    float v = (ci < 4) ? X[e0 * 4 * T_LEN + ci * T_LEN + t] * S[e0 * T_LEN + t]
                       : X[e1 * 4 * T_LEN + (ci - 4) * T_LEN + t] * S[e1 * T_LEN + t];
    xa[ci * T_LEN + t] = (f16)v;
  }
  for (int idx = lane; idx < 2 * T_LEN; idx += 32) {
    int ci = idx / T_LEN, t = idx - ci * T_LEN;
    sa[idx] = (f16)(ci == 0 ? S[e0 * T_LEN + t] : S[e1 * T_LEN + t]);
  }
  __syncthreads();

  int Tin = T_LEN;
  for (int k = 0; k < 6; ++k) {
    int Ci = c_att_in[k], Co = c_att_out[k], Cgi = c_attg_in[k];
    zpad(sa, Cgi * Tin, EBUF, lane);     // zero K-pad rows for im2col
    zpad(xa, Ci * Tin, EBUF, lane);
    conv_edge(w.gw[k], c_kp_attg[k], w.gb[k], sa, sb, Co, Tin, 2, lane);  // sigmoid
    conv_edge(w.aw[k], c_kp_att[k],  w.ab[k], xa, xb, Co, Tin, 1, lane);  // lrelu
    __syncthreads();
    int Tc = Tin - 4;
    if (k < 5) {
      if (c_pool[k] == 2) {
        int To = Tc >> 1;
        for (int idx = lane; idx < Co * To; idx += 32) {
          int c = idx / To, t = idx - c * To;
          float sv = fmaxf((float)sb[c * Tc + 2 * t], (float)sb[c * Tc + 2 * t + 1]);
          float xv = fmaxf((float)xb[c * Tc + 2 * t], (float)xb[c * Tc + 2 * t + 1]);
          sa[c * To + t] = (f16)sv;
          xa[c * To + t] = (f16)(xv * sv);   // X = X*S
        }
        Tin = To;
      } else {
        for (int idx = lane; idx < Co * Tc; idx += 32)
          xb[idx] = (f16)((float)xb[idx] * (float)sb[idx]);      // X = X*S
        f16* t1 = xa; xa = xb; xb = t1;
        t1 = sa; sa = sb; sb = t1;
        Tin = Tc;
      }
    } else {  // k==5, Tc==2: max over time, then X = X*S
      for (int c = lane; c < Co; c += 32) {
        float sv = fmaxf((float)sb[c * 2], (float)sb[c * 2 + 1]);
        float xv = fmaxf((float)xb[c * 2], (float)xb[c * 2 + 1]);
        xa[c] = (f16)(xv * sv);
      }
    }
    __syncthreads();
  }

  float p = 0.f;
  for (int c = lane; c < 128; c += 32) p += (float)xa[c] * w.linw[c];
  for (int off = 16; off; off >>= 1) p += __shfl_down(p, off);
  if (lane == 0) sim[e] = 1.f / (1.f + __expf(-(p + w.linb[0])));
}

// ---------------------------------------------------------------------------
// Node propagation: one wave per node, 6 gated conv layers via WMMA
// ---------------------------------------------------------------------------
#define PADB (32 * 69 + 64)   // padded im2col buffer + slack

struct NodePropW {
  const f16* pw[6]; const float* pb[6];
  const f16* gw[6]; const float* gb[6];
};

__global__ void __launch_bounds__(32)
node_prop_kernel(const float* __restrict__ Xagg, const float* __restrict__ Sagg,
                 float* __restrict__ Xbuf, float* __restrict__ Sbuf, NodePropW w) {
  __shared__ float bufA[32 * 65], bufB[32 * 65], bufC[32 * 65], bufD[32 * 65];
  __shared__ f16 padX[PADB], padS[PADB];
  int lane = threadIdx.x;
  int node = blockIdx.x;
  float* curX = bufA; float* outX = bufB;
  float* curS = bufC; float* outS = bufD;

  for (int i = lane; i < 4 * 65; i += 32) curX[i] = Xagg[node * 260 + i];
  for (int i = lane; i < 65; i += 32)     curS[i] = Sagg[node * 65 + i];
  __syncthreads();

  for (int k = 0; k < 6; ++k) {
    int Cx = c_prop[k], Cxo = c_prop[k + 1];
    int Cs = c_gate[k], Cso = c_gate[k + 1];
    // edge-pad X*scores and scores into f16 buffers (T 65 -> 69)
    for (int idx = lane; idx < Cx * 69; idx += 32) {
      int ci = idx / 69, tp = idx - ci * 69;
      int t = tp - 2; t = t < 0 ? 0 : (t > 64 ? 64 : t);
      float s = (Cs == 1) ? curS[t] : curS[ci * 65 + t];
      padX[ci * 69 + tp] = (f16)(curX[ci * 65 + t] * s);
    }
    for (int idx = lane; idx < Cs * 69; idx += 32) {
      int ci = idx / 69, tp = idx - ci * 69;
      int t = tp - 2; t = t < 0 ? 0 : (t > 64 ? 64 : t);
      padS[ci * 69 + tp] = (f16)curS[ci * 65 + t];
    }
    zpad(padX, Cx * 69, PADB, lane);
    zpad(padS, Cs * 69, PADB, lane);
    __syncthreads();
    conv_node(w.gw[k], c_kp_gate[k], w.gb[k], padS, outS, nullptr, Cso, 2, lane);
    conv_node(w.pw[k], c_kp_prop[k], w.pb[k], padX, outX,
              (Cx == Cxo) ? curX : nullptr, Cxo, 1, lane);
    __syncthreads();
    float* t1 = curX; curX = outX; outX = t1;
    t1 = curS; curS = outS; outS = t1;
  }
  for (int i = lane; i < 4 * 65; i += 32)
    Xbuf[node * 260 + i] = curX[i] + Xbuf[node * 260 + i];   // outer residual
  for (int i = lane; i < 65; i += 32)
    Sbuf[node * 65 + i] = curS[i];
}

// ---------------------------------------------------------------------------
// Generic f16 WMMA GEMM: C = act(A @ B(+bias)); one wave per 16x16 tile.
// Grid: (N/16, M/128); block = 8 waves sharing one N-strip (B tile in LDS).
// M%128==0, N%16==0, K%32==0, lda%8==0 (buffers padded by host).
// ---------------------------------------------------------------------------
struct GemmArgs {
  const f16* A; const f16* B; const float* bias;
  float* Cf; f16* Ch;
  int M, N, K, lda, ldb, ldc, act, transB;
};

__global__ void __launch_bounds__(256) gemm16_kernel(GemmArgs g) {
  __shared__ f16 bt[32 * 16];
  int wid = threadIdx.x >> 5, lane = threadIdx.x & 31;
  int ni = blockIdx.x;
  int mi = blockIdx.y * 8 + wid;
  const int m  = mi * 16 + (lane & 15);
  const int nn = ni * 16 + (lane & 15);
  v8f acc = {};

  if (g.transB) {  // B stored NxK row-major: fully contiguous per-lane loads
    for (int k0 = 0; k0 < g.K; k0 += 32) {
      int kb = k0 + ((lane & 16) ? 16 : 0);
      const v8h* ap8 = (const v8h*)(g.A + (size_t)m * g.lda + kb);
      v8h lo = ap8[0], hi = ap8[1];
      v16h a;
      #pragma unroll
      for (int i = 0; i < 8; ++i) { a[i] = lo[i]; a[i + 8] = hi[i]; }
      int kb2 = k0 + ((lane & 16) ? 16 : 0);
      const v8h* bp8 = (const v8h*)(g.B + (size_t)nn * g.ldb + kb2);
      v8h b0 = bp8[0], b1 = bp8[1];
      v16h b;
      #pragma unroll
      for (int i = 0; i < 8; ++i) { b[i] = b0[i]; b[i + 8] = b1[i]; }
      acc = WMMA16(a, b, acc);
    }
  } else {         // stage 32x16 B tile in LDS, shared by all 8 waves
    int kr = threadIdx.x >> 3, nc = (threadIdx.x & 7) << 1;
    for (int k0 = 0; k0 < g.K; k0 += 32) {
      __syncthreads();
      *(uint32_t*)(bt + kr * 16 + nc) =
          *(const uint32_t*)(g.B + (size_t)(k0 + kr) * g.ldb + ni * 16 + nc);
      __syncthreads();
      int kb = k0 + ((lane & 16) ? 16 : 0);
      const v8h* ap8 = (const v8h*)(g.A + (size_t)m * g.lda + kb);
      v8h lo = ap8[0], hi = ap8[1];
      v16h a;
      #pragma unroll
      for (int i = 0; i < 8; ++i) { a[i] = lo[i]; a[i + 8] = hi[i]; }
      int kb2 = (lane & 16) ? 16 : 0;
      v16h b;
      #pragma unroll
      for (int i = 0; i < 16; ++i) b[i] = bt[(kb2 + i) * 16 + (lane & 15)];
      acc = WMMA16(a, b, acc);
    }
  }

  float bz = g.bias ? g.bias[nn] : 0.f;
  #pragma unroll
  for (int v = 0; v < 8; ++v) {
    int mm = mi * 16 + v + ((lane & 16) ? 8 : 0);
    float x = actf(acc[v] + bz, g.act);
    if (g.Cf) g.Cf[(size_t)mm * g.ldc + nn] = x;
    if (g.Ch) g.Ch[(size_t)mm * g.ldc + nn] = (f16)x;
  }
}

// ---------------------------------------------------------------------------
// Small utility kernels
// ---------------------------------------------------------------------------
__global__ void cvt_f16_kernel(const float* s, f16* d, int n) {
  int i = blockIdx.x * blockDim.x + threadIdx.x;
  if (i < n) d[i] = (f16)s[i];
}
// convert + zero-pad conv weights: [Cout x Kd] -> [Mp x Kp]
__global__ void cvtpad_kernel(const float* s, f16* d, int Cout, int Kd, int Mp, int Kp) {
  int i = blockIdx.x * blockDim.x + threadIdx.x;
  if (i < Mp * Kp) {
    int m = i / Kp, k = i - m * Kp;
    d[i] = (m < Cout && k < Kd) ? (f16)s[m * Kd + k] : (f16)0.0f;
  }
}
__global__ void fill_f16_kernel(f16* d, int n) {
  int i = blockIdx.x * blockDim.x + threadIdx.x;
  if (i < n) d[i] = (f16)0.0f;
}
__global__ void scatter_kernel(const int* eidx, const float* sim, f16* A16) {
  int e = blockIdx.x * blockDim.x + threadIdx.x;
  if (e < N_EDGES) {
    int e0 = eidx[2 * e], e1 = eidx[2 * e + 1];
    f16 v = (f16)sim[e];
    A16[(size_t)e0 * N_NODES + e1] = v;
    A16[(size_t)e1 * N_NODES + e0] = v;
  }
}
__global__ void diag_kernel(f16* A16) {
  int i = blockIdx.x * blockDim.x + threadIdx.x;
  if (i < N_NODES) {
    size_t idx = (size_t)i * N_NODES + i;
    A16[idx] = (f16)((float)A16[idx] + 1.0f);   // att += eye (A[i,i]==1)
  }
}
__global__ void rowsum_kernel(const f16* A16, float* Dm) {
  __shared__ float red[256];
  int r = blockIdx.x;
  float s = 0.f;
  for (int j = threadIdx.x; j < N_NODES; j += 256)
    s += (float)A16[(size_t)r * N_NODES + j];
  red[threadIdx.x] = s; __syncthreads();
  for (int o = 128; o; o >>= 1) {
    if (threadIdx.x < o) red[threadIdx.x] += red[threadIdx.x + o];
    __syncthreads();
  }
  if (threadIdx.x == 0) Dm[r] = fmaxf(red[0] - EPS_K, 0.f) + EPS_K;
}
// Z = [X*scores (260 cols) | scores (65 cols)] as f16, ldb=336 (padded, pre-zeroed)
__global__ void zbuild_kernel(const float* Xb, const float* Sb, f16* Z) {
  int id = blockIdx.x * blockDim.x + threadIdx.x;
  if (id < N_NODES * 325) {
    int n = id / 325, c = id - n * 325;
    float v;
    if (c < 260) { int t = c % 65; v = Xb[n * 260 + c] * Sb[n * 65 + t]; }
    else         { v = Sb[n * 65 + (c - 260)]; }
    Z[(size_t)n * 336 + c] = (f16)v;
  }
}
// X = P_x / scales ; scores = P_s / Dm
__global__ void agg_kernel(const float* P, const float* Dm, float* Xagg, float* Sagg) {
  int id = blockIdx.x * blockDim.x + threadIdx.x;
  if (id < N_NODES * 65) {
    int n = id / 65, t = id - n * 65;
    float ps = P[(size_t)n * 336 + 260 + t];
    float sc = fmaxf(ps - EPS_K, 0.f) + EPS_K;
    #pragma unroll
    for (int d = 0; d < 4; ++d)
      Xagg[n * 260 + d * 65 + t] = P[(size_t)n * 336 + d * 65 + t] / sc;
    Sagg[n * 65 + t] = ps / Dm[n];
  }
}
__global__ void xfbuild_kernel(const float* Xb, f16* Xf16) {
  int id = blockIdx.x * blockDim.x + threadIdx.x;
  if (id < N_NODES * 260) {
    int n = id / 260, c = id - n * 260;
    Xf16[(size_t)n * 288 + c] = (f16)Xb[n * 260 + c];
  }
}
__global__ void norm_kernel(const float* Xf3, f16* Xn16, float* sq) {
  __shared__ float red[128];
  int r = blockIdx.x, t = threadIdx.x;
  float v = Xf3[(size_t)r * 128 + t];
  red[t] = v * v; __syncthreads();
  for (int o = 64; o; o >>= 1) {
    if (t < o) red[t] += red[t + o];
    __syncthreads();
  }
  float sum = red[0];
  float den = fmaxf(sqrtf(sum), 1e-12f);
  Xn16[(size_t)r * 128 + t] = (f16)(v / den);
  if (t == 0) sq[r] = sum / (den * den);
}
__global__ void dist_kernel(float* D, const float* sq) {
  int id = blockIdx.x * blockDim.x + threadIdx.x;
  if (id < N_NODES * N_NODES) {
    int i = id >> 11, j = id & 2047;
    float d2 = sq[i] + sq[j] - 2.f * D[id];
    D[id] = sqrtf(fmaxf(d2, 1e-12f));
  }
}

// ---------------------------------------------------------------------------
// Host orchestration
// ---------------------------------------------------------------------------
static const int H_ATT_IN[6]  = {8, 8, 16, 32, 64, 64};
static const int H_ATT_OUT[6] = {8, 16, 32, 64, 64, 128};
static const int H_ATTG_IN[6] = {2, 8, 16, 32, 64, 64};
static const int H_PROP[7]    = {4, 8, 16, 32, 32, 32, 4};
static const int H_GATE[7]    = {1, 8, 16, 32, 32, 32, 1};
static inline int rnd16(int x) { return (x + 15) & ~15; }
static inline int rnd32(int x) { return (x + 31) & ~31; }

extern "C" void kernel_launch(void* const* d_in, const int* in_sizes, int n_in,
                              void* d_out, int out_size, void* d_ws, size_t ws_size,
                              hipStream_t stream) {
  (void)in_sizes; (void)n_in; (void)out_size; (void)ws_size;
  auto cdiv = [](long a, long b) { return (int)((a + b - 1) / b); };

  // ---- inputs (pytree leaves: X, scores, A, edge_idx, then params sorted by key)
  const float* Xin = (const float*)d_in[0];
  const float* Sin = (const float*)d_in[1];
  const int*   eidx = (const int*)d_in[3];
  int p = 4;
  const float *att_b[6], *att_w[6], *attg_b[6], *attg_w[6];
  for (int k = 0; k < 6; ++k) att_b[k]  = (const float*)d_in[p++];
  for (int k = 0; k < 6; ++k) att_w[k]  = (const float*)d_in[p++];
  for (int k = 0; k < 6; ++k) attg_b[k] = (const float*)d_in[p++];
  for (int k = 0; k < 6; ++k) attg_w[k] = (const float*)d_in[p++];
  const float *gate_b[4][6], *gate_w[4][6];
  for (int l = 0; l < 4; ++l) for (int k = 0; k < 6; ++k) gate_b[l][k] = (const float*)d_in[p++];
  for (int l = 0; l < 4; ++l) for (int k = 0; k < 6; ++k) gate_w[l][k] = (const float*)d_in[p++];
  const float* lin_b = (const float*)d_in[p++];
  const float* lin_w = (const float*)d_in[p++];
  const float *prop_b[4][6], *prop_w[4][6];
  for (int l = 0; l < 4; ++l) for (int k = 0; k < 6; ++k) prop_b[l][k] = (const float*)d_in[p++];
  for (int l = 0; l < 4; ++l) for (int k = 0; k < 6; ++k) prop_w[l][k] = (const float*)d_in[p++];
  const float *trip_b[3], *trip_w[3];
  for (int k = 0; k < 3; ++k) trip_b[k] = (const float*)d_in[p++];
  for (int k = 0; k < 3; ++k) trip_w[k] = (const float*)d_in[p++];

  // ---- workspace carve (~25 MB)
  char* wptr = (char*)d_ws;
  auto alloc = [&](size_t bytes) -> void* {
    void* r = (void*)wptr;
    wptr += (bytes + 255) & ~(size_t)255;
    return r;
  };
  f16*   A16  = (f16*)alloc((size_t)N_NODES * N_NODES * 2);
  f16*   Z16  = (f16*)alloc((size_t)N_NODES * 336 * 2);
  float* Pbuf = (float*)alloc((size_t)N_NODES * 336 * 4);
  float* Xbuf = (float*)alloc((size_t)N_NODES * 260 * 4);
  float* Sbuf = (float*)alloc((size_t)N_NODES * 65 * 4);
  float* Xagg = (float*)alloc((size_t)N_NODES * 260 * 4);
  float* Sagg = (float*)alloc((size_t)N_NODES * 65 * 4);
  float* simb = (float*)alloc((size_t)N_EDGES * 4);
  float* Dm   = (float*)alloc((size_t)N_NODES * 4);
  f16*   Xf16 = (f16*)alloc((size_t)N_NODES * 288 * 2);
  f16*   T1   = (f16*)alloc((size_t)N_NODES * 256 * 2);
  f16*   T2   = (f16*)alloc((size_t)N_NODES * 256 * 2);
  float* Xf3  = (float*)alloc((size_t)N_NODES * 128 * 4);
  f16*   Xn16 = (f16*)alloc((size_t)N_NODES * 128 * 2);
  float* sq   = (float*)alloc((size_t)N_NODES * 4);
  f16 *attw16[6], *attgw16[6], *gatew16[4][6], *propw16[4][6];
  for (int k = 0; k < 6; ++k)
    attw16[k]  = (f16*)alloc((size_t)rnd16(H_ATT_OUT[k]) * rnd32(H_ATT_IN[k] * 5) * 2);
  for (int k = 0; k < 6; ++k)
    attgw16[k] = (f16*)alloc((size_t)rnd16(H_ATT_OUT[k]) * rnd32(H_ATTG_IN[k] * 5) * 2);
  for (int l = 0; l < 4; ++l) for (int k = 0; k < 6; ++k)
    gatew16[l][k] = (f16*)alloc((size_t)rnd16(H_GATE[k + 1]) * rnd32(H_GATE[k] * 5) * 2);
  for (int l = 0; l < 4; ++l) for (int k = 0; k < 6; ++k)
    propw16[l][k] = (f16*)alloc((size_t)rnd16(H_PROP[k + 1]) * rnd32(H_PROP[k] * 5) * 2);
  f16* W1p = (f16*)alloc((size_t)288 * 256 * 2);   // (260->288 pad) x 256
  f16* W2  = (f16*)alloc((size_t)256 * 256 * 2);
  f16* W3  = (f16*)alloc((size_t)256 * 128 * 2);

  auto cvt = [&](const float* s, f16* d, int n) {
    cvt_f16_kernel<<<cdiv(n, 256), 256, 0, stream>>>(s, d, n);
  };
  auto cvtpad = [&](const float* s, f16* d, int Cout, int Cin) {
    int Kd = Cin * 5, Mp = rnd16(Cout), Kp = rnd32(Kd);
    cvtpad_kernel<<<cdiv(Mp * Kp, 256), 256, 0, stream>>>(s, d, Cout, Kd, Mp, Kp);
  };
  // ---- weight conversions (deterministic, every call)
  for (int k = 0; k < 6; ++k) cvtpad(att_w[k],  attw16[k],  H_ATT_OUT[k], H_ATT_IN[k]);
  for (int k = 0; k < 6; ++k) cvtpad(attg_w[k], attgw16[k], H_ATT_OUT[k], H_ATTG_IN[k]);
  for (int l = 0; l < 4; ++l) for (int k = 0; k < 6; ++k) {
    cvtpad(gate_w[l][k], gatew16[l][k], H_GATE[k + 1], H_GATE[k]);
    cvtpad(prop_w[l][k], propw16[l][k], H_PROP[k + 1], H_PROP[k]);
  }
  fill_f16_kernel<<<cdiv(288 * 256, 256), 256, 0, stream>>>(W1p, 288 * 256);
  cvt(trip_w[0], W1p, 260 * 256);
  cvt(trip_w[1], W2, 256 * 256);
  cvt(trip_w[2], W3, 256 * 128);
  fill_f16_kernel<<<cdiv((long)N_NODES * 336, 256), 256, 0, stream>>>(Z16, N_NODES * 336);
  fill_f16_kernel<<<cdiv((long)N_NODES * 288, 256), 256, 0, stream>>>(Xf16, N_NODES * 288);

  hipMemcpyAsync(Xbuf, Xin, (size_t)N_NODES * 260 * 4, hipMemcpyDeviceToDevice, stream);
  hipMemcpyAsync(Sbuf, Sin, (size_t)N_NODES * 65 * 4, hipMemcpyDeviceToDevice, stream);

  auto gemm = [&](const f16* A, const f16* B, const float* bias, float* Cf, f16* Ch,
                  int M, int N, int K, int lda, int ldb, int ldc, int act, int transB) {
    GemmArgs ga{A, B, bias, Cf, Ch, M, N, K, lda, ldb, ldc, act, transB};
    dim3 grid(N / 16, M / 128);
    gemm16_kernel<<<grid, 256, 0, stream>>>(ga);
  };

  EdgeSimW ew;
  for (int k = 0; k < 6; ++k) {
    ew.aw[k] = attw16[k];  ew.ab[k] = att_b[k];
    ew.gw[k] = attgw16[k]; ew.gb[k] = attg_b[k];
  }
  ew.linw = lin_w; ew.linb = lin_b;

  // ---- 4 graph layers
  for (int l = 0; l < 4; ++l) {
    edge_sim_kernel<<<N_EDGES / ESIM_WAVES, 128, 0, stream>>>(Xbuf, Sbuf, eidx, simb, ew);
    fill_f16_kernel<<<cdiv((long)N_NODES * N_NODES, 256), 256, 0, stream>>>(A16, N_NODES * N_NODES);
    scatter_kernel<<<cdiv(N_EDGES, 256), 256, 0, stream>>>(eidx, simb, A16);
    diag_kernel<<<cdiv(N_NODES, 256), 256, 0, stream>>>(A16);
    rowsum_kernel<<<N_NODES, 256, 0, stream>>>(A16, Dm);
    zbuild_kernel<<<cdiv((long)N_NODES * 325, 256), 256, 0, stream>>>(Xbuf, Sbuf, Z16);
    // P = A_upd @ [X*s | s]  (2048 x 336, K = 2048)
    gemm(A16, Z16, nullptr, Pbuf, nullptr, N_NODES, 336, N_NODES, N_NODES, 336, 336, 0, 0);
    agg_kernel<<<cdiv((long)N_NODES * 65, 256), 256, 0, stream>>>(Pbuf, Dm, Xagg, Sagg);
    NodePropW nw;
    for (int k = 0; k < 6; ++k) {
      nw.pw[k] = propw16[l][k]; nw.pb[k] = prop_b[l][k];
      nw.gw[k] = gatew16[l][k]; nw.gb[k] = gate_b[l][k];
    }
    node_prop_kernel<<<N_NODES, 32, 0, stream>>>(Xagg, Sagg, Xbuf, Sbuf, nw);
  }

  // ---- triplet MLP + pairwise distance (only final layer's dist is returned)
  xfbuild_kernel<<<cdiv((long)N_NODES * 260, 256), 256, 0, stream>>>(Xbuf, Xf16);
  gemm(Xf16, W1p, trip_b[0], nullptr, T1, N_NODES, 256, 288, 288, 256, 256, 1, 0);
  gemm(T1,   W2,  trip_b[1], nullptr, T2, N_NODES, 256, 256, 256, 256, 256, 1, 0);
  gemm(T2,   W3,  trip_b[2], Xf3, nullptr, N_NODES, 128, 256, 256, 128, 128, 0, 0);
  norm_kernel<<<N_NODES, 128, 0, stream>>>(Xf3, Xn16, sq);
  // G = Xn @ Xn^T into d_out, then in-place distance transform
  gemm(Xn16, Xn16, nullptr, (float*)d_out, nullptr, N_NODES, N_NODES, 128, 128, 128, N_NODES, 0, 1);
  dist_kernel<<<cdiv((long)N_NODES * N_NODES, 256), 256, 0, stream>>>((float*)d_out, sq);
}
